// AutoEncoderModular_20229295964349
// MI455X (gfx1250) — matrix operand order repaired and checked by reading, compiled, not run on hardware
//
#include <hip/hip_runtime.h>

#define GN   14
#define CPG  8
#define CH   112
#define NEXP 16

typedef __attribute__((ext_vector_type(16))) _Float16 v16h;
typedef __attribute__((ext_vector_type(8)))  float    v8f;

typedef union {
    v16h         v;
    unsigned int u[8];
    uint4        q[2];
} frag_t;

// workspace (ints): [0..15] counts | [16..32] offsets | [40..55] cursors | [64..64+n) perm

__global__ void k_zero(int* ws) {
    if (threadIdx.x < 64) ws[threadIdx.x] = 0;
}

__global__ void k_count(const int* __restrict__ keys, int* ws, int n) {
    int i = blockIdx.x * blockDim.x + threadIdx.x;
    if (i < n) atomicAdd(&ws[keys[i] & 15], 1);
}

__global__ void k_scan(int* ws) {
    if (threadIdx.x == 0) {
        int acc = 0;
        for (int e = 0; e < NEXP; ++e) { ws[16 + e] = acc; acc += ws[e]; }
        ws[32] = acc;
    }
}

__global__ void k_scatter(const int* __restrict__ keys, int* ws, int n) {
    int i = blockIdx.x * blockDim.x + threadIdx.x;
    if (i < n) {
        int k = keys[i] & 15;
        int pos = ws[16 + k] + atomicAdd(&ws[40 + k], 1);
        ws[64 + pos] = i;
    }
}

// One workgroup = 16 samples that all use expert e.
__global__ __launch_bounds__(256)
void k_moe_ae(const float* __restrict__ x,     // [n][14][8][8]
              const float* __restrict__ mask9, // [3][3]
              const float* __restrict__ ecw,   // [E][112][9]
              const float* __restrict__ edw,   // [E][14][8][128]
              const float* __restrict__ edb,   // [E][14][8]
              const float* __restrict__ ddw,   // [E][14][128][8]
              const float* __restrict__ ddb,   // [E][14][128]
              const float* __restrict__ tcw,   // [E][112][9]
              const float* __restrict__ tcb,   // [E][14]
              float* __restrict__ out,         // [n][14][64]
              const int* __restrict__ ws)
{
    __shared__ __align__(16) _Float16 featY[16 * GN * 128];   // 57344 B: feat, reused as y
    __shared__ __align__(16) _Float16 zbuf[16 * GN * CPG];    // 3584 B
    __shared__ float    wsm[CH * 9];                          // enc conv (masked) / tconv w
    __shared__ int      sidx[16];
    __shared__ int      shdr[4];

    const int tid  = threadIdx.x;
    const int lane = tid & 31;
    const int wave = tid >> 5;
    const int hi   = lane >> 4;
    const int lm   = lane & 15;

    // map blockIdx -> (expert, local tile)
    if (tid == 0) {
        int t = (int)blockIdx.x, acc = 0, ee = -1, loc = 0;
        for (int e = 0; e < NEXP; ++e) {
            int te = (ws[e] + 15) >> 4;
            if (t < acc + te) { ee = e; loc = t - acc; break; }
            acc += te;
        }
        int mm = 0;
        if (ee >= 0) { mm = ws[ee] - loc * 16; if (mm > 16) mm = 16; }
        shdr[0] = ee; shdr[1] = loc; shdr[2] = mm;
    }
    __syncthreads();
    const int e = shdr[0];
    if (e < 0) return;
    const int loc = shdr[1];
    const int m   = shdr[2];

    if (tid < 16) sidx[tid] = (tid < m) ? ws[64 + ws[16 + e] + loc * 16 + tid] : -1;

    // stage masked encoder conv weights
    for (int idx = tid; idx < CH * 9; idx += 256)
        wsm[idx] = ecw[e * CH * 9 + idx] * mask9[idx % 9];
    __syncthreads();

    // ---- Stage A: conv + relu + maxpool -> feat (f16 in LDS) ----
    for (int it = tid; it < 16 * GN * CPG * 16; it += 256) {
        int pp = it & 15;  int t2 = it >> 4;
        int cc = t2 & 7;   t2 >>= 3;
        int g  = t2 % GN;  int i = t2 / GN;
        int pr = pp >> 2,  pc = pp & 3;
        float mx = 0.f;
        int si = sidx[i];
        if (si >= 0) {
            const float* xin = x + (si * GN + g) * 64;
            const float* w   = &wsm[(g * CPG + cc) * 9];
            #pragma unroll
            for (int dr = 0; dr < 2; ++dr)
            #pragma unroll
            for (int dc = 0; dc < 2; ++dc) {
                int r = pr * 2 + dr, c = pc * 2 + dc;
                float s = 0.f;
                #pragma unroll
                for (int kh = 0; kh < 3; ++kh) {
                    int rr = r + kh - 1;
                    if (rr < 0 || rr > 7) continue;
                    #pragma unroll
                    for (int kw = 0; kw < 3; ++kw) {
                        int qq = c + kw - 1;
                        if (qq < 0 || qq > 7) continue;
                        s = fmaf(w[kh * 3 + kw], xin[rr * 8 + qq], s);
                    }
                }
                mx = fmaxf(mx, s);   // relu folded: mx starts at 0
            }
        }
        featY[(i * GN + g) * 128 + cc * 16 + pr * 4 + pc] = (_Float16)mx;
    }
    __syncthreads();

    // ---- Stage B: encoder dense 128->8 via WMMA (per group) ----
    // A: lane=sample(lm); LDS b128 loads. B: lane=o (dup rows for lm>=8, cols 8..15 discarded).
    #pragma unroll
    for (int gi = 0; gi < 2; ++gi) {
        const int g = wave + gi * 8;
        if (g < GN) {
            v8f acc = {};
            const uint4*  fr4 = (const uint4*)(featY + (lm * GN + g) * 128);
            const float4* wr4 = (const float4*)(edw + ((e * GN + g) * CPG + (lm & 7)) * 128);
            #pragma unroll
            for (int kc = 0; kc < 4; ++kc) {
                frag_t fa, fb;
                fa.q[0] = fr4[kc * 4 + hi * 2];       // K = kc*32 + hi*16 .. +7
                fa.q[1] = fr4[kc * 4 + hi * 2 + 1];   // K = kc*32 + hi*16 + 8 .. +15
                float4 w0 = wr4[kc * 8 + hi * 4 + 0];   // K = kc*32 + hi*16 .. +15
                float4 w1 = wr4[kc * 8 + hi * 4 + 1];
                float4 w2 = wr4[kc * 8 + hi * 4 + 2];
                float4 w3 = wr4[kc * 8 + hi * 4 + 3];
                fb.v[0]  = (_Float16)w0.x; fb.v[1]  = (_Float16)w0.y;
                fb.v[2]  = (_Float16)w0.z; fb.v[3]  = (_Float16)w0.w;
                fb.v[4]  = (_Float16)w1.x; fb.v[5]  = (_Float16)w1.y;
                fb.v[6]  = (_Float16)w1.z; fb.v[7]  = (_Float16)w1.w;
                fb.v[8]  = (_Float16)w2.x; fb.v[9]  = (_Float16)w2.y;
                fb.v[10] = (_Float16)w2.z; fb.v[11] = (_Float16)w2.w;
                fb.v[12] = (_Float16)w3.x; fb.v[13] = (_Float16)w3.y;
                fb.v[14] = (_Float16)w3.z; fb.v[15] = (_Float16)w3.w;
                acc = __builtin_amdgcn_wmma_f32_16x16x32_f16(false, fa.v, false, fb.v,
                                                             (short)0, acc, false, false);
            }
            if (lm < 8) {
                float bias = edb[(e * GN + g) * CPG + lm];
                #pragma unroll
                for (int r = 0; r < 8; ++r) {
                    int samp = r + hi * 8;
                    zbuf[(samp * GN + g) * CPG + lm] = (_Float16)fmaxf(acc[r] + bias, 0.f);
                }
            }
        }
    }
    __syncthreads();

    // ---- Stage C: decoder dense 8->128 via WMMA (K zero-padded to 32); y -> featY ----
    #pragma unroll
    for (int gi = 0; gi < 2; ++gi) {
        const int g = wave + gi * 8;
        if (g < GN) {
            frag_t fa;
            uint4 qz = *(const uint4*)(zbuf + (lm * GN + g) * CPG);  // K=0..7
            qz.x = hi ? 0u : qz.x;  qz.y = hi ? 0u : qz.y;           // hi lanes cover K=8..15 -> 0
            qz.z = hi ? 0u : qz.z;  qz.w = hi ? 0u : qz.w;
            fa.q[0] = qz;
            fa.u[4] = 0u; fa.u[5] = 0u; fa.u[6] = 0u; fa.u[7] = 0u;  // K=16..31 -> 0
            #pragma unroll
            for (int t = 0; t < 8; ++t) {
                int f = t * 16 + lm;
                const float4* wc4 = (const float4*)(ddw + ((e * GN + g) * 128 + f) * CPG);
                float4 w0 = wc4[0], w1 = wc4[1];                     // o = 0..7
                frag_t fb;
                fb.v[0] = (_Float16)w0.x; fb.v[1] = (_Float16)w0.y;
                fb.v[2] = (_Float16)w0.z; fb.v[3] = (_Float16)w0.w;
                fb.v[4] = (_Float16)w1.x; fb.v[5] = (_Float16)w1.y;
                fb.v[6] = (_Float16)w1.z; fb.v[7] = (_Float16)w1.w;
                fb.u[0] = hi ? 0u : fb.u[0];  fb.u[1] = hi ? 0u : fb.u[1];
                fb.u[2] = hi ? 0u : fb.u[2];  fb.u[3] = hi ? 0u : fb.u[3];
                fb.u[4] = 0u; fb.u[5] = 0u; fb.u[6] = 0u; fb.u[7] = 0u;
                v8f d = {};
                d = __builtin_amdgcn_wmma_f32_16x16x32_f16(false, fa.v, false, fb.v,
                                                           (short)0, d, false, false);
                float bias = ddb[(e * GN + g) * 128 + f];
                #pragma unroll
                for (int r = 0; r < 8; ++r) {
                    int samp = r + hi * 8;
                    featY[(samp * GN + g) * 128 + f] = (_Float16)fmaxf(d[r] + bias, 0.f);
                }
            }
        }
    }
    __syncthreads();

    // restage weights: decoder transposed-conv
    for (int idx = tid; idx < CH * 9; idx += 256)
        wsm[idx] = tcw[e * CH * 9 + idx];
    __syncthreads();

    // ---- Stage D: ConvTranspose2d(8->1 per group, s=2, p=1, op=1) + bias + relu ----
    for (int it = tid; it < 16 * GN * 64; it += 256) {
        int pq = it & 63; int t2 = it >> 6;
        int g = t2 % GN;  int i = t2 / GN;
        int si = sidx[i];
        if (si < 0) continue;
        int p = pq >> 3, q = pq & 7;
        float acc = tcb[e * GN + g];
        #pragma unroll
        for (int kh = 0; kh < 3; ++kh) {
            int ph = p + 1 - kh;
            if (ph < 0 || ph > 7 || (ph & 1)) continue;
            int ih = ph >> 1;
            #pragma unroll
            for (int kw = 0; kw < 3; ++kw) {
                int pw = q + 1 - kw;
                if (pw < 0 || pw > 7 || (pw & 1)) continue;
                int iw = pw >> 1;
                #pragma unroll
                for (int cc = 0; cc < 8; ++cc) {
                    acc = fmaf(wsm[(g * CPG + cc) * 9 + kh * 3 + kw],
                               (float)featY[(i * GN + g) * 128 + cc * 16 + ih * 4 + iw], acc);
                }
            }
        }
        __builtin_nontemporal_store(fmaxf(acc, 0.f), &out[(si * GN + g) * 64 + pq]);
    }
}

extern "C" void kernel_launch(void* const* d_in, const int* in_sizes, int n_in,
                              void* d_out, int out_size, void* d_ws, size_t ws_size,
                              hipStream_t stream) {
    const float* x    = (const float*)d_in[0];
    const int*   keys = (const int*)  d_in[1];
    const float* mask = (const float*)d_in[2];
    const float* ecw  = (const float*)d_in[3];
    const float* edw  = (const float*)d_in[4];
    const float* edb  = (const float*)d_in[5];
    const float* ddw  = (const float*)d_in[6];
    const float* ddb  = (const float*)d_in[7];
    const float* tcw  = (const float*)d_in[8];
    const float* tcb  = (const float*)d_in[9];
    float* out = (float*)d_out;
    int*   ws  = (int*)d_ws;
    const int n = in_sizes[0] / (GN * 64);

    k_zero<<<1, 64, 0, stream>>>(ws);
    k_count<<<(n + 255) / 256, 256, 0, stream>>>(keys, ws, n);
    k_scan<<<1, 32, 0, stream>>>(ws);
    k_scatter<<<(n + 255) / 256, 256, 0, stream>>>(keys, ws, n);

    const int tiles = (n + 15) / 16 + NEXP;  // per-expert partial-tile slack
    k_moe_ae<<<tiles, 256, 0, stream>>>(x, mask, ecw, edw, edb, ddw, ddb, tcw, tcb, out, ws);
}
